// LaGAT_4629974745754
// MI455X (gfx1250) — matrix operand order repaired and checked by reading, not compiled, not run on hardware
//
#include <hip/hip_runtime.h>
#include <hip/hip_bf16.h>

// ---------------- problem constants (match reference) ----------------
#define N_NODES   4096
#define B_GRAPH   64
#define NPG       64
#define HID       64
#define POS       32
#define INIT_EMB  32
#define NLAYER    3
#define LDH       72     // row stride (halves) for row-major A-side LDS tiles
                         // 72*2 = 144 B = 9*16 B -> A-chunk loads stay 16B-aligned

typedef __attribute__((ext_vector_type(16))) _Float16 v16h;
typedef __attribute__((ext_vector_type(8)))  float    v8f;

// ---------------------------------------------------------------------
// CDNA5 WMMA fragment helpers (wave32), A/B in contiguous k order (k = 16*(lane/16) + elem).
//
// A-matrix 16x32 f16 (row-major LDS): lanes 0-15 hold row m = lane&15;
//   VGPR v in 0..7 -> K = 2v,2v+1 (contiguous, matches swizzled B);
//   lanes 16-31: same rows, K += 16.
// Per lane this is two contiguous 16B chunks -> compiler emits 2x ds_load_b128.
__device__ __forceinline__ v16h load_a_frag(const _Float16* X, int mt, int kt, int lane) {
  const int m   = mt * 16 + (lane & 15);
  const int grp = lane >> 4;
  v16h a;
#pragma unroll
  for (int v = 0; v < 8; ++v) {
    const int k0 = kt * 32 + 2 * v + 16 * grp;
    a[2 * v]     = X[m * LDH + k0];
    a[2 * v + 1] = X[m * LDH + k0 + 1];
  }
  return a;
}

// B-matrix 32x16 f16: lanes 0-15 hold col n = nt*16 + (lane&15) with
// K = kt*32 + 0..15 (2 consecutive K per VGPR); lanes 16-31 same cols, K += 16.
// We PRE-SWIZZLE B operands into fragment-major storage: slot (kt,nt,lane)
// holds its 16 halves contiguously (32B, 32B-aligned) -> one v16h LDS load
// (2x ds_load_b128) per fragment instead of 16 strided ds_load_u16.
__device__ __forceinline__ int b_slot(int kt, int nt, int lane) {
  return (((kt << 2) | nt) * 32 + lane) * 16;   // units: halves
}

__device__ __forceinline__ v16h load_b_frag_sw(const _Float16* Wf, int kt, int nt, int lane) {
  return *(const v16h*)(Wf + b_slot(kt, nt, lane));
}

// C/D f32 16x16: element (m,n) in VGPR (m&7); m = mt*16 + v + 8*(lane>>4),
// n = nt*16 + (lane&15).

// ---------------------------------------------------------------------
// Kernel 1: feat = concat(feat_node, embed[idx]) @ transform  [4096 x 64]
// One block per graph (64 rows), 128 threads = 4 waves; wave w owns rows
// w*16..w*16+15.  Writes feat to workspace and output layer 0.
__global__ void lagat_feat_kernel(const float* __restrict__ feat_node,
                                  const int*   __restrict__ idx,
                                  const float* __restrict__ transform,
                                  const float* __restrict__ embed,
                                  float* __restrict__ featws,
                                  float* __restrict__ out) {
  __shared__ __align__(32) _Float16 xinh[64 * LDH];   // A: row-major
  __shared__ __align__(32) _Float16 trf[4096];        // B: fragment-major
  const int g    = blockIdx.x;
  const int t    = threadIdx.x;
  const int lane = t & 31;
  const int wave = t >> 5;
  const int hlf  = lane >> 4;

  // stage A (gathered input features) row-major
  for (int e = t; e < 4096; e += 128) {
    const int i = e >> 6, k = e & 63;
    const int node = g * NPG + i;
    float v;
    if (k < POS) v = feat_node[node * POS + k];
    else         v = embed[(size_t)idx[node] * INIT_EMB + (k - POS)];
    xinh[i * LDH + k] = (_Float16)v;
  }
  // stage B (transform) directly into swizzled fragment slots
  for (int s = t; s < 256; s += 128) {
    const int frag = s >> 5, ls = s & 31;
    const int kt = frag >> 2, nt = frag & 3;
    const int grp = ls >> 4, n = nt * 16 + (ls & 15);
    v16h vb;
#pragma unroll
    for (int v = 0; v < 8; ++v) {
      const int k = kt * 32 + 2 * v + 16 * grp;
      vb[2 * v]     = (_Float16)transform[k * HID + n];
      vb[2 * v + 1] = (_Float16)transform[(k + 1) * HID + n];
    }
    *(v16h*)(trf + s * 16) = vb;
  }
  __syncthreads();

  v8f acc[4];
#pragma unroll
  for (int nt = 0; nt < 4; ++nt)
#pragma unroll
    for (int v = 0; v < 8; ++v) acc[nt][v] = 0.0f;

#pragma unroll
  for (int kt = 0; kt < 2; ++kt) {
    const v16h a = load_a_frag(xinh, wave, kt, lane);
#pragma unroll
    for (int nt = 0; nt < 4; ++nt) {
      const v16h b = load_b_frag_sw(trf, kt, nt, lane);
      acc[nt] = __builtin_amdgcn_wmma_f32_16x16x32_f16(
          false, a, false, b, (short)0, acc[nt], false, false);
    }
  }

#pragma unroll
  for (int nt = 0; nt < 4; ++nt)
#pragma unroll
    for (int v = 0; v < 8; ++v) {
      const int m = wave * 16 + v + 8 * hlf;
      const int n = nt * 16 + (lane & 15);
      const int node = g * NPG + m;
      const float val = acc[nt][v];
      featws[(size_t)node * HID + n] = val;
      out[(size_t)node * ((NLAYER + 1) * HID) + n] = val;   // layer 0
    }
}

// ---------------------------------------------------------------------
// Kernel 2: per-graph LaGAT layers.  One block per graph, 128 thr = 4 waves.
// All 64x64 tiles resident in LDS (f16); GEMMs via v_wmma_f32_16x16x32_f16;
// LayerNorm fully in-register with wave32 shuffles over the C-fragment layout.
__global__ void lagat_layers_kernel(const float* __restrict__ adj,
                                    const int*   __restrict__ head_ids,
                                    const int*   __restrict__ tail_ids,
                                    const float* __restrict__ featws,
                                    const float* __restrict__ w1g,
                                    const float* __restrict__ b1g,
                                    const float* __restrict__ w2g,
                                    const float* __restrict__ b2g,
                                    const float* __restrict__ gamma1,
                                    const float* __restrict__ beta1,
                                    const float* __restrict__ gamma2,
                                    const float* __restrict__ beta2,
                                    float* __restrict__ out) {
  __shared__ __align__(32) _Float16 adjh[64 * LDH];   // A: adjacency (0/1 exact in f16)
  __shared__ __align__(32) _Float16 xworw[64 * LDH];  // union: wei (B, swizzled, first 4096 halves)
                                                      //        / xin (A, row-major) — disjoint lifetimes
  __shared__ __align__(32) _Float16 afh[64 * LDH];    // all_feat_head
  __shared__ __align__(32) _Float16 aftl[64 * LDH];   // all_feat_tail
  __shared__ __align__(32) _Float16 w1f[4096];        // B: fragment-major
  __shared__ __align__(32) _Float16 w2f[4096];        // B: fragment-major
  __shared__ float fhop[NLAYER * 64];                 // head hop flags (0/1)
  __shared__ float thop[NLAYER * 64];                 // tail hop flags
  __shared__ float deg[64];
  __shared__ float q1[64];                            // feat[head] (headQ)
  __shared__ float q2[64];                            // feat[tail] (tailQ)
  __shared__ float cor[64];

  const int g    = blockIdx.x;
  const int t    = threadIdx.x;
  const int lane = t & 31;
  const int wave = t >> 5;
  const int hlf  = lane >> 4;
  const int base = g * NPG;
  const int headL = head_ids[g] - base;   // local head index (== 0)
  const int tailL = tail_ids[g] - base;   // local tail index (== 1)

  // hint the DMA-ish prefetcher at our working set (global_prefetch_b8)
  __builtin_prefetch(adj + (size_t)(base + (t >> 1)) * N_NODES + base, 0, 0);
  __builtin_prefetch(featws + (size_t)(base + (t >> 1)) * HID, 0, 0);

  // ---- stage adjacency block + initial features (row-major) ----
  for (int e = t; e < 4096; e += 128) {
    const int i = e >> 6, j = e & 63;
    adjh[i * LDH + j] = (_Float16)adj[(size_t)(base + i) * N_NODES + base + j];
    const float fe = featws[(size_t)(base + i) * HID + j];
    afh[i * LDH + j]  = (_Float16)fe;
    aftl[i * LDH + j] = (_Float16)fe;
  }
  // ---- stage w1/w2 straight into swizzled B fragment slots ----
  for (int s = t; s < 256; s += 128) {
    const int frag = s >> 5, ls = s & 31;
    const int kt = frag >> 2, nt = frag & 3;
    const int grp = ls >> 4, n = nt * 16 + (ls & 15);
    v16h vb1, vb2;
#pragma unroll
    for (int v = 0; v < 8; ++v) {
      const int k = kt * 32 + 2 * v + 16 * grp;
      vb1[2 * v]     = (_Float16)w1g[k * HID + n];
      vb1[2 * v + 1] = (_Float16)w1g[(k + 1) * HID + n];
      vb2[2 * v]     = (_Float16)w2g[k * HID + n];
      vb2[2 * v + 1] = (_Float16)w2g[(k + 1) * HID + n];
    }
    *(v16h*)(w1f + s * 16) = vb1;
    *(v16h*)(w2f + s * 16) = vb2;
  }
  __syncthreads();

  // ---- degree + queries + hop-0 flags ----
  if (t < 64) {
    float s = 0.0f;
    for (int j = 0; j < 64; ++j) s += (float)adjh[t * LDH + j];
    deg[t] = s;
    q1[t] = featws[(size_t)(base + headL) * HID + t];
    q2[t] = featws[(size_t)(base + tailL) * HID + t];
    fhop[t] = (t == headL) ? 1.0f : 0.0f;
    thop[t] = (t == tailL) ? 1.0f : 0.0f;
  }
  __syncthreads();

  // ---- hop flags: rows head/tail of binarized adjacency powers ----
  for (int k = 1; k < NLAYER; ++k) {
    if (t < 64) {
      float sh = 0.0f, st = 0.0f;
      for (int j = 0; j < 64; ++j) {
        const float a = (float)adjh[j * LDH + t];
        sh += fhop[(k - 1) * 64 + j] * a;
        st += thop[(k - 1) * 64 + j] * a;
      }
      fhop[k * 64 + t] = (sh > 0.0f || t == headL) ? 1.0f : 0.0f;
      thop[k * 64 + t] = (st > 0.0f || t == tailL) ? 1.0f : 0.0f;
    }
    __syncthreads();
  }

  // ---- layer loop: hop = NLAYER-1 .. 0 ----
  for (int hop = NLAYER - 1; hop >= 0; --hop) {
    for (int side = 0; side < 2; ++side) {
      const float* f     = (side == 0 ? fhop : thop) + hop * 64;
      const float* q     = (side == 0 ? q2 : q1);     // side1 queries headQ
      _Float16* af       = (side == 0 ? afh : aftl);
      const _Float16* wf = (side == 0 ? w1f : w2f);
      const float* bias  = (side == 0 ? b1g : b2g);
      const float* gam   = (side == 0 ? gamma1 : gamma2);
      const float* bet   = (side == 0 ? beta1 : beta2);

      // (a) cor[i] = | <query, all_feat[i]> |
      if (t < 64) {
        float s = 0.0f;
        for (int h = 0; h < 64; ++h) s += q[h] * (float)af[t * LDH + h];
        cor[t] = fabsf(s);
      }
      __syncthreads();

      // (b) wei = cor * all_feat, emitted directly into swizzled B slots
      for (int s = t; s < 256; s += 128) {
        const int frag = s >> 5, ls = s & 31;
        const int kt = frag >> 2, nt = frag & 3;
        const int grp = ls >> 4, n = nt * 16 + (ls & 15);
        v16h vb;
#pragma unroll
        for (int v = 0; v < 8; ++v) {
          const int k = kt * 32 + 2 * v + 16 * grp;
          vb[2 * v]     = (_Float16)(cor[k]     * (float)af[k * LDH + n]);
          vb[2 * v + 1] = (_Float16)(cor[k + 1] * (float)af[(k + 1) * LDH + n]);
        }
        *(v16h*)(xworw + s * 16) = vb;   // wei lives in the union buffer
      }
      __syncthreads();

      // (c) up_raw = adj @ wei   (WMMA)
      v8f acc[4];
#pragma unroll
      for (int nt = 0; nt < 4; ++nt)
#pragma unroll
        for (int v = 0; v < 8; ++v) acc[nt][v] = 0.0f;
#pragma unroll
      for (int kt = 0; kt < 2; ++kt) {
        const v16h a = load_a_frag(adjh, wave, kt, lane);
#pragma unroll
        for (int nt = 0; nt < 4; ++nt) {
          const v16h b = load_b_frag_sw(xworw, kt, nt, lane);
          acc[nt] = __builtin_amdgcn_wmma_f32_16x16x32_f16(
              false, a, false, b, (short)0, acc[nt], false, false);
        }
      }

      // (d) mask scale + leaky_relu + /num_nei, all in registers
      float xreg[4][8];
#pragma unroll
      for (int nt = 0; nt < 4; ++nt)
#pragma unroll
        for (int v = 0; v < 8; ++v) {
          const int m = wave * 16 + v + 8 * hlf;
          float u = f[m] * acc[nt][v];
          u = (u > 0.0f) ? u : 0.01f * u;
          xreg[nt][v] = u / (f[m] * deg[m] + 1e-8f);
        }

      // (e) LayerNorm per row via shuffles within the 16-lane row group,
      //     then fuse the aggregator input: xin = f*feat + LN(x)
#pragma unroll
      for (int v = 0; v < 8; ++v) {
        float s  = xreg[0][v] + xreg[1][v] + xreg[2][v] + xreg[3][v];
        float ss = xreg[0][v] * xreg[0][v] + xreg[1][v] * xreg[1][v] +
                   xreg[2][v] * xreg[2][v] + xreg[3][v] * xreg[3][v];
#pragma unroll
        for (int off = 8; off >= 1; off >>= 1) {
          s  += __shfl_xor(s, off, 32);
          ss += __shfl_xor(ss, off, 32);
        }
        const float mu  = s * (1.0f / 64.0f);
        const float var = ss * (1.0f / 64.0f) - mu * mu;
        const float inv = rsqrtf(var + 1e-5f);
        const int m  = wave * 16 + v + 8 * hlf;
        const float fm = f[m];
#pragma unroll
        for (int nt = 0; nt < 4; ++nt) {
          const int n = nt * 16 + (lane & 15);
          const float xn = gam[n] * (xreg[nt][v] - mu) * inv + bet[n];
          const float fe = featws[(size_t)(base + m) * HID + n];
          xreg[nt][v] = fm * fe + xn;   // aggregator input (entity + neighbor)
        }
      }
      __syncthreads();   // all waves finished reading wei as B operand

      // (f) write xin (f16, row-major A layout) into the union buffer
#pragma unroll
      for (int nt = 0; nt < 4; ++nt)
#pragma unroll
        for (int v = 0; v < 8; ++v) {
          const int m = wave * 16 + v + 8 * hlf;
          const int n = nt * 16 + (lane & 15);
          xworw[m * LDH + n] = (_Float16)xreg[nt][v];
        }
      __syncthreads();

      // (g) agg = relu(xin @ W + b)   (WMMA, bias preloaded into C)
      v8f acc2[4];
#pragma unroll
      for (int nt = 0; nt < 4; ++nt) {
        const float bv = bias[nt * 16 + (lane & 15)];
#pragma unroll
        for (int v = 0; v < 8; ++v) acc2[nt][v] = bv;
      }
#pragma unroll
      for (int kt = 0; kt < 2; ++kt) {
        const v16h a = load_a_frag(xworw, wave, kt, lane);
#pragma unroll
        for (int nt = 0; nt < 4; ++nt) {
          const v16h b = load_b_frag_sw(wf, kt, nt, lane);
          acc2[nt] = __builtin_amdgcn_wmma_f32_16x16x32_f16(
              false, a, false, b, (short)0, acc2[nt], false, false);
        }
      }

      // (h) predicated update: all_feat = where(f>0, relu(agg), all_feat)
#pragma unroll
      for (int nt = 0; nt < 4; ++nt)
#pragma unroll
        for (int v = 0; v < 8; ++v) {
          const int m = wave * 16 + v + 8 * hlf;
          const int n = nt * 16 + (lane & 15);
          if (f[m] > 0.0f) {
            const float r = acc2[nt][v];
            af[m * LDH + n] = (_Float16)(r > 0.0f ? r : 0.0f);
          }
        }
      __syncthreads();
    }  // side

    // ---- emit layer (NLAYER - hop): all_feat_head + all_feat_tail ----
    const int li = NLAYER - hop;
    for (int e = t; e < 4096; e += 128) {
      const int i = e >> 6, h = e & 63;
      out[(size_t)(base + i) * ((NLAYER + 1) * HID) + li * HID + h] =
          (float)afh[i * LDH + h] + (float)aftl[i * LDH + h];
    }
    __syncthreads();
  }
}

// ---------------------------------------------------------------------
extern "C" void kernel_launch(void* const* d_in, const int* in_sizes, int n_in,
                              void* d_out, int out_size, void* d_ws, size_t ws_size,
                              hipStream_t stream) {
  (void)in_sizes; (void)n_in; (void)out_size; (void)ws_size;
  const float* adj       = (const float*)d_in[0];
  const float* feat_node = (const float*)d_in[1];
  const int*   idx       = (const int*)d_in[2];
  const int*   head_ids  = (const int*)d_in[3];
  const int*   tail_ids  = (const int*)d_in[4];
  // d_in[5] = graph_indices (block structure: node n belongs to graph n/64)
  const float* transform = (const float*)d_in[6];
  const float* embed     = (const float*)d_in[7];
  const float* w1        = (const float*)d_in[8];
  const float* b1        = (const float*)d_in[9];
  const float* w2        = (const float*)d_in[10];
  const float* b2        = (const float*)d_in[11];
  const float* gamma1    = (const float*)d_in[12];
  const float* beta1     = (const float*)d_in[13];
  const float* gamma2    = (const float*)d_in[14];
  const float* beta2     = (const float*)d_in[15];
  float* out    = (float*)d_out;
  float* featws = (float*)d_ws;   // 4096*64 f32 = 1 MB scratch for feat

  lagat_feat_kernel<<<dim3(B_GRAPH), dim3(128), 0, stream>>>(
      feat_node, idx, transform, embed, featws, out);
  lagat_layers_kernel<<<dim3(B_GRAPH), dim3(128), 0, stream>>>(
      adj, head_ids, tail_ids, featws, w1, b1, w2, b2,
      gamma1, beta1, gamma2, beta2, out);
}
